// SetEncoderPointNetSp_35424890257454
// MI455X (gfx1250) — matrix operand order repaired and checked by reading, not compiled, not run on hardware
//
#include <hip/hip_runtime.h>
#include <hip/hip_bf16.h>

// ---------------------------------------------------------------------------
// SetEncoderPointNetSp for MI455X (gfx1250, wave32, WMMA)
//   z      = x @ W1 + b1            (E x 256)   -- bf16 WMMA, f32 accum
//   z_max  = segment_max(z, vid)    (V x 256)   -- u32 atomic max on monotonic
//                                                  float encoding, L2-resident
//   out    = concat(x, z_max[vid]) @ W2 (E x 128) -- bf16 WMMA, f32 accum
// ---------------------------------------------------------------------------

typedef __attribute__((ext_vector_type(16))) __bf16 v16bf;
typedef __attribute__((ext_vector_type(8)))  float  v8f;

#define DIN   128
#define DMID  256
#define DOUT  128
#define DCAT  (DIN + DMID)   // 384
#define VNUM  50000

#define LDX   (DIN  + 8)     // x tile row stride (bf16)      : 136 (272B, 16B aligned)
#define LDK1  (DIN  + 8)     // W1^T column stride (bf16)     : 136
#define LDA   (DCAT + 8)     // concat tile row stride (bf16) : 392 (784B, 16B aligned)
#define LDK2  (DCAT + 8)     // W2^T column stride (bf16)     : 392

// Monotonic float<->uint mapping: order-preserving, so float max == u32 max.
__device__ __forceinline__ unsigned enc_f32(float f) {
  unsigned b = __float_as_uint(f);
  return (b & 0x80000000u) ? ~b : (b | 0x80000000u);
}
__device__ __forceinline__ float dec_f32(unsigned u) {
  unsigned b = (u & 0x80000000u) ? (u & 0x7fffffffu) : ~u;
  return __uint_as_float(b);
}
#define ENC_NEG_INF 0x007FFFFFu   // enc(-inf): identity of segment_max

// A fragment, 16x32 bf16 (row-major tile in LDS), per CDNA5 ISA 7.12.2:
//   lanes 0-15: M=lane, VGPR0..3 hold K=0..7, VGPR4..7 hold K=16..23
//   lanes16-31: same M rows, K=8..15 / K=24..31
__device__ __forceinline__ v16bf load_frag_A(const __bf16* __restrict__ tile,
                                             int ld, int kbase, int lane) {
  const int m = lane & 15, half = lane >> 4;
  const __bf16* row = tile + m * ld + kbase + half * 8;
  v16bf a;
#pragma unroll
  for (int e = 0; e < 16; ++e)
    a[e] = row[((e >> 3) << 4) + (e & 7)];   // two 16B contiguous runs
  return a;
}

// B fragment, 32x16 bf16, weights stored COLUMN-major in LDS (W^T):
//   lane owns column (ncol + lane&15); K = half*8 + e (e<8), 16 + half*8 + (e-8) (e>=8)
__device__ __forceinline__ v16bf load_frag_B(const __bf16* __restrict__ wt,
                                             int ldk, int kbase, int ncol, int lane) {
  const int n = lane & 15, half = lane >> 4;
  const __bf16* col = wt + (ncol + n) * ldk + kbase + half * 8;
  v16bf b;
#pragma unroll
  for (int e = 0; e < 16; ++e) b[e] = col[((e >> 3) << 4) + (e & 7)];  // two 16B contiguous runs
  return b;
}

__global__ void init_zmax(unsigned* __restrict__ zmax, long long count) {
  long long i = (long long)blockIdx.x * blockDim.x + threadIdx.x;
  if (i < count) zmax[i] = ENC_NEG_INF;
}

// ---------------------------------------------------------------------------
// Kernel 1: z = x@W1 + b1, scatter-max into zmax (encoded u32, L2-resident)
// Block: 256 threads (8 waves), 128 rows x 256 cols; wave w owns cols [32w,32w+32)
// ---------------------------------------------------------------------------
__global__ __launch_bounds__(256) void gemm1_scatter_max(
    const float* __restrict__ x, const int* __restrict__ vid,
    const float* __restrict__ W1, const float* __restrict__ b1,
    unsigned* __restrict__ zmax, int E)
{
  __shared__ __bf16 sW1t[DMID * LDK1];   // W1^T, 256 cols x 136  (~68 KB)
  __shared__ __bf16 sX[128 * LDX];       // x tile               (~35 KB)
  __shared__ float  sB1[DMID];
  __shared__ int    sVid[128];

  const int tid   = threadIdx.x;
  const int rbase = blockIdx.x * 128;

  // stage W1 (global row-major, coalesced) -> LDS column-major bf16
  for (int idx = tid; idx < DIN * DMID; idx += 256) {
    int k = idx >> 8, c = idx & (DMID - 1);
    sW1t[c * LDK1 + k] = (__bf16)W1[idx];
  }
  if (tid < DMID) sB1[tid] = b1[tid];
  if (tid < 128) {
    int gr = rbase + tid;
    sVid[tid] = (gr < E) ? vid[gr] : 0;
  }
  // stage x tile (float4 coalesced) -> bf16 LDS
  for (int idx = tid; idx < 128 * (DIN / 4); idx += 256) {
    int r = idx >> 5, c4 = (idx & 31) << 2;
    int gr = rbase + r;
    float4 t = (gr < E) ? ((const float4*)x)[(long long)gr * (DIN / 4) + (idx & 31)]
                        : make_float4(0.f, 0.f, 0.f, 0.f);
    __bf16* d = &sX[r * LDX + c4];
    d[0] = (__bf16)t.x; d[1] = (__bf16)t.y; d[2] = (__bf16)t.z; d[3] = (__bf16)t.w;
  }
  __syncthreads();

  const int lane = tid & 31, wave = tid >> 5;
  const int half = lane >> 4, n = lane & 15;
  const int nb0  = wave * 32;

  for (int rg = 0; rg < 8; ++rg) {
    const int m0 = rg * 16;
    v8f acc0 = {}, acc1 = {};
#pragma unroll
    for (int kb = 0; kb < DIN; kb += 32) {
      v16bf a  = load_frag_A(&sX[m0 * LDX], LDX, kb, lane);
      v16bf b0 = load_frag_B(sW1t, LDK1, kb, nb0,      lane);
      v16bf b1f= load_frag_B(sW1t, LDK1, kb, nb0 + 16, lane);
      acc0 = __builtin_amdgcn_wmma_f32_16x16x32_bf16(false, a, false, b0,
                                                     (short)0, acc0, false, false);
      acc1 = __builtin_amdgcn_wmma_f32_16x16x32_bf16(false, a, false, b1f,
                                                     (short)0, acc1, false, false);
    }
    // C layout: element i -> (M = i + 8*half, N = n); bias-add then atomic max
#pragma unroll
    for (int i = 0; i < 8; ++i) {
      int r  = m0 + i + 8 * half;
      int gr = rbase + r;
      if (gr < E) {
        unsigned* dst = zmax + (long long)sVid[r] * DMID;
        float v0 = acc0[i] + sB1[nb0 + n];
        float v1 = acc1[i] + sB1[nb0 + 16 + n];
        atomicMax(dst + (nb0 + n),      enc_f32(v0));   // global_atomic_max_u32
        atomicMax(dst + (nb0 + 16 + n), enc_f32(v1));
      }
    }
  }
}

// ---------------------------------------------------------------------------
// Kernel 2: out = concat(x, zmax[vid]) @ W2
// Block: 256 threads (8 waves), 64 rows x 128 cols; wave w owns cols [16w,16w+16)
// ---------------------------------------------------------------------------
__global__ __launch_bounds__(256) void gemm2_gather(
    const float* __restrict__ x, const int* __restrict__ vid,
    const float* __restrict__ W2, const unsigned* __restrict__ zmax,
    float* __restrict__ out, int E)
{
  __shared__ __bf16 sA[64 * LDA];        // concat tile (~49 KB)
  __shared__ __bf16 sW2t[DOUT * LDK2];   // W2^T        (~98 KB)

  const int tid   = threadIdx.x;
  const int rbase = blockIdx.x * 64;

  for (int idx = tid; idx < DCAT * DOUT; idx += 256) {
    int k = idx >> 7, c = idx & (DOUT - 1);
    sW2t[c * LDK2 + k] = (__bf16)W2[idx];
  }
  // x part of the concat tile
  for (int idx = tid; idx < 64 * (DIN / 4); idx += 256) {
    int r = idx >> 5, c4 = (idx & 31) << 2;
    int gr = rbase + r;
    float4 t = (gr < E) ? ((const float4*)x)[(long long)gr * (DIN / 4) + (idx & 31)]
                        : make_float4(0.f, 0.f, 0.f, 0.f);
    __bf16* d = &sA[r * LDA + c4];
    d[0] = (__bf16)t.x; d[1] = (__bf16)t.y; d[2] = (__bf16)t.z; d[3] = (__bf16)t.w;
  }
  // gathered zmax part (zmax is 51MB -> L2 hits)
  for (int idx = tid; idx < 64 * DMID; idx += 256) {
    int r = idx >> 8, c = idx & (DMID - 1);
    int gr = rbase + r;
    float z = 0.f;
    if (gr < E) z = dec_f32(zmax[(long long)vid[gr] * DMID + c]);
    sA[r * LDA + DIN + c] = (__bf16)z;
  }
  __syncthreads();

  const int lane = tid & 31, wave = tid >> 5;
  const int half = lane >> 4, n = lane & 15;
  const int nb   = wave * 16;

  for (int rg = 0; rg < 4; ++rg) {
    const int m0 = rg * 16;
    v8f acc = {};
#pragma unroll
    for (int kb = 0; kb < DCAT; kb += 32) {
      v16bf a = load_frag_A(&sA[m0 * LDA], LDA, kb, lane);
      v16bf b = load_frag_B(sW2t, LDK2, kb, nb, lane);
      acc = __builtin_amdgcn_wmma_f32_16x16x32_bf16(false, a, false, b,
                                                    (short)0, acc, false, false);
    }
#pragma unroll
    for (int i = 0; i < 8; ++i) {
      int gr = rbase + m0 + i + 8 * half;
      if (gr < E) out[(long long)gr * DOUT + nb + n] = acc[i];
    }
  }
}

// ---------------------------------------------------------------------------
extern "C" void kernel_launch(void* const* d_in, const int* in_sizes, int n_in,
                              void* d_out, int out_size, void* d_ws, size_t ws_size,
                              hipStream_t stream) {
  const float* x   = (const float*)d_in[0];
  const int*   vid = (const int*)  d_in[1];
  const float* W1  = (const float*)d_in[2];
  const float* b1  = (const float*)d_in[3];
  const float* W2  = (const float*)d_in[4];
  float*       out = (float*)d_out;

  const int E = in_sizes[1];                       // edge count from vertex_id
  unsigned* zmax = (unsigned*)d_ws;                // V*DMID u32 = 51.2 MB scratch
  const long long zcount = (long long)VNUM * DMID;

  init_zmax<<<(unsigned)((zcount + 1023) / 1024), 1024, 0, stream>>>(zmax, zcount);
  gemm1_scatter_max<<<(E + 127) / 128, 256, 0, stream>>>(x, vid, W1, b1, zmax, E);
  gemm2_gather<<<(E + 63) / 64, 256, 0, stream>>>(x, vid, W2, zmax, out, E);
}
